// pos_prediction_18588618457701
// MI455X (gfx1250) — compile-verified, matrix operand order checked by reading, not run on hardware
//
#include <hip/hip_runtime.h>
#include <hip/hip_bf16.h>
#include <cstdint>

// ---------------------------------------------------------------------------
// eSCN-style equivariant block for MI455X (gfx1250), bf16 WMMA everywhere.
// Weights pre-packed into d_ws as bf16 WMMA B-fragments (contiguous 32B/lane);
// intermediate activations that feed a B operand are scattered into the same
// fragment layout in LDS so no strided LDS reads remain in the GEMM loops.
// ---------------------------------------------------------------------------

typedef __attribute__((ext_vector_type(16))) __bf16 v16bf;
typedef __attribute__((ext_vector_type(8)))  float  v8f;

#define NT 16   // samples per block
#define NC 25   // SH coefficients
#define CH 64

// ws layout (bf16 elements)
#define WS_SCW  0        // scalar_w  64x64   ->  8 frags
#define WS_L1W  4096     // lin1_w  5x64x64   -> 40 frags
#define WS_GW1  24576    // grid_w1  64x128   -> 16 frags
#define WS_GW2  32768    // grid_w2 128x128   -> 32 frags
#define WS_GW3  49152    // grid_w3 128x64    -> 16 frags
#define WS_L2W  57344    // lin2_w  5x64x64   -> 40 frags
#define WS_TOTAL 77824   // elements (155,648 bytes)

__constant__ int DEGC[25] = {0,
                             1,1,1,
                             2,2,2,2,2,
                             3,3,3,3,3,3,3,
                             4,4,4,4,4,4,4,4,4};

// K index mapping inside a 16-element bf16 fragment (CDNA5 ISA 7.12.2):
// element e -> K = (e&7) + (e>=8 ? 16 : 0)   (plus +8 for lanes 16..31)
__device__ __forceinline__ int kmap(int e) { return (e & 7) | ((e >> 3) << 4); }
// inverse pieces: for K value k ->  e = (k&7)|(((k>>4)&1)<<3),  lane_hi = (k>>3)&1
__device__ __forceinline__ int k2e(int k)  { return (k & 7) | (((k >> 4) & 1) << 3); }
__device__ __forceinline__ int k2hi(int k) { return (k >> 3) & 1; }

// A fragment: A is M(16) x K(32), row-major with given element stride.
__device__ __forceinline__ v16bf load_a_lds(const __bf16* base, int stride, int lane) {
  const __bf16* p = base + (lane & 15) * stride + ((lane >> 4) << 3);
  v16bf a;
#pragma unroll
  for (int e = 0; e < 16; ++e) a[e] = p[kmap(e)];
  return a;
}

// B fragment from fragment-layout storage (LDS or global): 32B contiguous/lane.
__device__ __forceinline__ v16bf load_b_frag(const __bf16* region, int frag, int lane) {
  const v16bf* p = (const v16bf*)region;
  return p[frag * 32 + lane];
}

__device__ __forceinline__ v8f wmma_bf16(v16bf a, v16bf b, v8f c) {
  return __builtin_amdgcn_wmma_f32_16x16x32_bf16(false, a, false, b, (short)0, c,
                                                 false, false);
}

__device__ __forceinline__ float silu(float v) { return v / (1.0f + __expf(-v)); }

__device__ __forceinline__ float red16(float v) {
  v += __shfl_xor(v, 1, 32);
  v += __shfl_xor(v, 2, 32);
  v += __shfl_xor(v, 4, 32);
  v += __shfl_xor(v, 8, 32);
  return v;
}

// ======================= weight pre-pack kernel ============================
__device__ __forceinline__ void pack_one(const float* src, __bf16* dst,
                                         int K, int N, int rel) {
  const int lane = rel & 31;
  const int frag = rel >> 5;
  const int ntiles = N >> 4;
  const int nt = frag % ntiles;
  const int kt = frag / ntiles;
  __bf16* d = dst + (size_t)(frag * 32 + lane) * 16;
  const int kbase = kt * 32 + ((lane >> 4) << 3);
  const int n = nt * 16 + (lane & 15);
#pragma unroll
  for (int e = 0; e < 16; ++e) d[e] = (__bf16)src[(kbase + kmap(e)) * N + n];
}

__global__ __launch_bounds__(256) void pack_weights_kernel(
    const float* __restrict__ scw, const float* __restrict__ l1w,
    const float* __restrict__ gw1, const float* __restrict__ gw2,
    const float* __restrict__ gw3, const float* __restrict__ l2w,
    __bf16* __restrict__ ws) {
  const int idx = blockIdx.x * 256 + threadIdx.x;   // lane-row id, 4864 total
  if (idx < 256) {
    pack_one(scw, ws + WS_SCW, 64, 64, idx);
  } else if (idx < 1536) {
    const int rel = idx - 256;
    const int mtx = rel >> 8;
    pack_one(l1w + mtx * 4096, ws + WS_L1W + mtx * 4096, 64, 64, rel & 255);
  } else if (idx < 2048) {
    pack_one(gw1, ws + WS_GW1, 64, 128, idx - 1536);
  } else if (idx < 3072) {
    pack_one(gw2, ws + WS_GW2, 128, 128, idx - 2048);
  } else if (idx < 3584) {
    pack_one(gw3, ws + WS_GW3, 128, 64, idx - 3072);
  } else if (idx < 4864) {
    const int rel = idx - 3584;
    const int mtx = rel >> 8;
    pack_one(l2w + mtx * 4096, ws + WS_L2W + mtx * 4096, 64, 64, rel & 255);
  }
}

// ============================ main kernel ==================================
__global__ __launch_bounds__(256) void escn_block_kernel(
    const float* __restrict__ x,        // [N][25][64]
    const float* __restrict__ nl0w,     // [64]
    const float* __restrict__ nl0b,     // [64]
    const float* __restrict__ naff,     // [4][64]
    const float* __restrict__ l1b,      // [64]
    const float* __restrict__ scb,      // [64]
    const float* __restrict__ l2b,      // [64]
    const float* __restrict__ tg,       // [64][25]
    const __bf16* __restrict__ ws,      // packed weights
    float* __restrict__ out,            // [N][25][64]
    int N)
{
  // --- LDS ---
  // sPool: sXn[25][16][64] in stage 0/1, then grid activations + sGB.
  __shared__ __align__(64) unsigned char sPool[NC * NT * CH * 2];     // 51200 B
  // sHPool per sample (2048 bf16 = 4KB):
  //   stage1..forward-projection: B-fragment layout [4 frags][32 lanes][16]
  //   back-projection..lin2:      row-major [32 m][64 h]
  __shared__ __align__(64) __bf16 sHPool[NT * 2048];                  // 65536 B
  __shared__ __bf16 sGate[NT][CH];                                    //  2048 B
  __shared__ __bf16 sTgF[64][32];                                     //  4096 B (A[s][m])
  __shared__ __bf16 sTgB[32][64];                                     //  4096 B (A[m][s])

  __bf16 (*sXn)[NT][CH] = (__bf16(*)[NT][CH])sPool;                   // [25][16][64]
  __bf16 (*sG )[64]     = (__bf16(*)[64])sPool;                       // [64][64]
  __bf16 (*sT1)[128]    = (__bf16(*)[128])(sPool + 8192);             // [64][128]
  __bf16 (*sT2)[128]    = (__bf16(*)[128])(sPool + 8192 + 16384);     // [64][128]
  __bf16* sGB           = (__bf16*)(sPool + 40960);                   // 8 frags (8KB)
  __bf16 (*sHrm)[32][CH] = (__bf16(*)[32][CH])sHPool;                 // row-major view

  const int tid  = threadIdx.x;
  const int wave = tid >> 5;
  const int lane = tid & 31;
  const int n0   = blockIdx.x * NT;

  // ======================= stage 0: normalize into sXn ======================
  {
    const int s  = tid >> 4;
    const int cg = tid & 15;
    const int nclamp = min(n0 + s, N - 1);
    const float* xp = x + (size_t)nclamp * NC * CH;

    float v0[4];
    float sum = 0.f, sq = 0.f;
#pragma unroll
    for (int j = 0; j < 4; ++j) {
      v0[j] = xp[cg * 4 + j];
      sum += v0[j];
      sq  += v0[j] * v0[j];
    }
    sum = red16(sum);
    sq  = red16(sq);
    const float mu   = sum * (1.0f / 64.0f);
    const float var  = sq * (1.0f / 64.0f) - mu * mu;
    const float istd = rsqrtf(var + 1e-5f);
#pragma unroll
    for (int j = 0; j < 4; ++j) {
      const int c = cg * 4 + j;
      sXn[0][s][c] = (__bf16)(((v0[j] - mu) * istd) * nl0w[c] + nl0b[c]);
    }

    float part = 0.f;
    for (int m = 1; m < NC; ++m) {
      const int l = DEGC[m];
      const float bw = 0.25f / (float)(2 * l + 1);
#pragma unroll
      for (int j = 0; j < 4; ++j) {
        const float v = xp[m * CH + cg * 4 + j];
        part += bw * v * v;
      }
    }
    part = red16(part);
    const float inv = rsqrtf(part * (1.0f / 64.0f) + 1e-5f);
    for (int m = 1; m < NC; ++m) {
      const int l = DEGC[m];
#pragma unroll
      for (int j = 0; j < 4; ++j) {
        const int c = cg * 4 + j;
        sXn[m][s][c] = (__bf16)(xp[m * CH + c] * inv * naff[(l - 1) * CH + c]);
      }
    }
  }

  // zero all of sHPool (gives zero K-padding m=25..31 in the fragments)
  {
    float4 z; z.x = z.y = z.z = z.w = 0.0f;
    float4* p = (float4*)sHPool;
    for (int i = tid; i < NT * 2048 * 2 / 16; i += 256) p[i] = z;
  }
  // stage to_grid_mat into LDS, forward + backward, K-padded
  for (int i = tid; i < 64 * 32; i += 256) {
    const int s = i >> 5, m = i & 31;
    const __bf16 v = (m < NC) ? (__bf16)tg[s * NC + m] : (__bf16)0.0f;
    sTgF[s][m] = v;
    sTgB[m][s] = v;
  }
  __syncthreads();

  // =============== stage 1: gating (jobs 0..3) + lin1 (jobs 4..103) =========
  for (int job = wave; job < 104; job += 8) {
    if (job < 4) {
      const int nb = job * 16;
      v8f acc = {};
#pragma unroll
      for (int kt = 0; kt < 2; ++kt) {
        v16bf a = load_a_lds(&sXn[0][0][0] + kt * 32, CH, lane);
        v16bf b = load_b_frag(ws + WS_SCW, kt * 4 + job, lane);
        acc = wmma_bf16(a, b, acc);
      }
      const int col = lane & 15, rb = (lane >> 4) << 3;
      const float sb = scb[nb + col];
#pragma unroll
      for (int r = 0; r < 8; ++r)
        sGate[rb + r][nb + col] = (__bf16)silu(acc[r] + sb);
    } else {
      const int j  = job - 4;
      const int m  = j >> 2;              // K index of the next GEMM
      const int nt = j & 3;
      const int nb = nt * 16;
      const __bf16* W = ws + WS_L1W + DEGC[m] * 4096;
      v8f acc = {};
#pragma unroll
      for (int kt = 0; kt < 2; ++kt) {
        v16bf a = load_a_lds(&sXn[m][0][0] + kt * 32, CH, lane);
        v16bf b = load_b_frag(W, kt * 4 + nt, lane);
        acc = wmma_bf16(a, b, acc);
      }
      const int col = lane & 15, rb = (lane >> 4) << 3;
      const float bias = l1b[nb + col] * ((m == 0) ? 1.0f : 0.0f);
      // scatter into per-sample B-fragment layout for the to-grid projection
      const int tlane = (lane & 15) | (k2hi(m) << 4);
      const int e     = k2e(m);
#pragma unroll
      for (int r = 0; r < 8; ++r) {
        const int s = rb + r;
        sHPool[s * 2048 + nt * 512 + tlane * 16 + e] = (__bf16)(acc[r] + bias);
      }
    }
  }
  __syncthreads();

  // ==================== stage 2: grid MLP, one sample at a time =============
  for (int s = 0; s < NT; ++s) {
    // g = to_grid @ h[s]   (64 x 32 x 64); B from per-sample fragment layout
    for (int job = wave; job < 16; job += 8) {
      const int mt = job >> 2, nt = job & 3, nb = nt * 16;
      v16bf a = load_a_lds(&sTgF[mt * 16][0], 32, lane);
      v16bf b = load_b_frag(sHPool + s * 2048, nt, lane);
      v8f acc = {};
      acc = wmma_bf16(a, b, acc);
      const int col = lane & 15, rb = (lane >> 4) << 3;
#pragma unroll
      for (int r = 0; r < 8; ++r) sG[mt * 16 + rb + r][nb + col] = (__bf16)acc[r];
    }
    __syncthreads();

    // t1 = silu(g @ W1)   (64 x 64 x 128)
    for (int job = wave; job < 32; job += 8) {
      const int mt = job >> 3, nt = job & 7, nb = nt * 16;
      v8f acc = {};
#pragma unroll
      for (int kt = 0; kt < 2; ++kt) {
        v16bf a = load_a_lds(&sG[mt * 16][kt * 32], 64, lane);
        v16bf b = load_b_frag(ws + WS_GW1, kt * 8 + nt, lane);
        acc = wmma_bf16(a, b, acc);
      }
      const int col = lane & 15, rb = (lane >> 4) << 3;
#pragma unroll
      for (int r = 0; r < 8; ++r)
        sT1[mt * 16 + rb + r][nb + col] = (__bf16)silu(acc[r]);
    }
    __syncthreads();

    // t2 = silu(t1 @ W2)  (64 x 128 x 128)
    for (int job = wave; job < 32; job += 8) {
      const int mt = job >> 3, nt = job & 7, nb = nt * 16;
      v8f acc = {};
#pragma unroll
      for (int kt = 0; kt < 4; ++kt) {
        v16bf a = load_a_lds(&sT1[mt * 16][kt * 32], 128, lane);
        v16bf b = load_b_frag(ws + WS_GW2, kt * 8 + nt, lane);
        acc = wmma_bf16(a, b, acc);
      }
      const int col = lane & 15, rb = (lane >> 4) << 3;
#pragma unroll
      for (int r = 0; r < 8; ++r)
        sT2[mt * 16 + rb + r][nb + col] = (__bf16)silu(acc[r]);
    }
    __syncthreads();

    // g3 = t2 @ W3 (64 x 128 x 64): scatter into fragment layout sGB
    for (int job = wave; job < 16; job += 8) {
      const int mt = job >> 2, nt = job & 3;
      v8f acc = {};
#pragma unroll
      for (int kt = 0; kt < 4; ++kt) {
        v16bf a = load_a_lds(&sT2[mt * 16][kt * 32], 128, lane);
        v16bf b = load_b_frag(ws + WS_GW3, kt * 4 + nt, lane);
        acc = wmma_bf16(a, b, acc);
      }
      const int col = lane & 15, rb = (lane >> 4) << 3;
#pragma unroll
      for (int r = 0; r < 8; ++r) {
        const int gs = mt * 16 + rb + r;            // K index of back-projection
        const int kt2 = gs >> 5;
        const int tlane = (lane & 15) | (k2hi(gs) << 4);
        sGB[(kt2 * 4 + nt) * 512 + tlane * 16 + k2e(gs)] = (__bf16)acc[r];
      }
    }
    __syncthreads();

    // h2 = to_grid^T @ g3 (32 x 64 x 64); write rows m=1..24 row-major into
    // the (now dead) per-sample fragment region of sHPool.
    for (int job = wave; job < 8; job += 8) {
      const int mt = job >> 2, nt = job & 3, nb = nt * 16;
      v8f acc = {};
#pragma unroll
      for (int kt = 0; kt < 2; ++kt) {
        v16bf a = load_a_lds(&sTgB[mt * 16][kt * 32], 64, lane);
        v16bf b = load_b_frag(sGB, kt * 4 + nt, lane);
        acc = wmma_bf16(a, b, acc);
      }
      const int col = lane & 15, rb = (lane >> 4) << 3;
#pragma unroll
      for (int r = 0; r < 8; ++r) {
        const int m = mt * 16 + rb + r;
        if (m >= 1 && m < NC) sHrm[s][m][nb + col] = (__bf16)acc[r];
      }
    }
    __syncthreads();
  }

  // row m=0 of the lin2 input <- gating activations
  for (int i = tid; i < NT * CH; i += 256)
    sHrm[i >> 6][0][i & 63] = sGate[i >> 6][i & 63];
  __syncthreads();

  // ===================== stage 3: lin2 per-degree, write out ================
  for (int job = wave; job < 100; job += 8) {
    const int m  = job >> 2;
    const int nt = job & 3;
    const int nb = nt * 16;
    const __bf16* W = ws + WS_L2W + DEGC[m] * 4096;
    v8f acc = {};
#pragma unroll
    for (int kt = 0; kt < 2; ++kt) {
      v16bf a = load_a_lds(&sHrm[0][m][0] + kt * 32, 32 * CH, lane);  // sample stride
      v16bf b = load_b_frag(W, kt * 4 + nt, lane);
      acc = wmma_bf16(a, b, acc);
    }
    const int col = lane & 15, rb = (lane >> 4) << 3;
    const float bias = l2b[nb + col] * ((m == 0) ? 1.0f : 0.0f);
#pragma unroll
    for (int r = 0; r < 8; ++r) {
      const int n = n0 + rb + r;
      if (n < N) out[(size_t)n * NC * CH + m * CH + nb + col] = acc[r] + bias;
    }
  }
}

extern "C" void kernel_launch(void* const* d_in, const int* in_sizes, int n_in,
                              void* d_out, int out_size, void* d_ws, size_t ws_size,
                              hipStream_t stream) {
  (void)n_in; (void)ws_size; (void)out_size;
  const float* x    = (const float*)d_in[0];
  const float* nl0w = (const float*)d_in[1];
  const float* nl0b = (const float*)d_in[2];
  const float* naff = (const float*)d_in[3];
  const float* l1w  = (const float*)d_in[4];
  const float* l1b  = (const float*)d_in[5];
  const float* scw  = (const float*)d_in[6];
  const float* scb  = (const float*)d_in[7];
  const float* gw1  = (const float*)d_in[8];
  const float* gw2  = (const float*)d_in[9];
  const float* gw3  = (const float*)d_in[10];
  const float* l2w  = (const float*)d_in[11];
  const float* l2b  = (const float*)d_in[12];
  const float* tg   = (const float*)d_in[13];
  float* out = (float*)d_out;
  __bf16* ws = (__bf16*)d_ws;

  pack_weights_kernel<<<19, 256, 0, stream>>>(scw, l1w, gw1, gw2, gw3, l2w, ws);

  const int N = in_sizes[0] / (NC * CH);
  const int nblocks = (N + NT - 1) / NT;
  escn_block_kernel<<<nblocks, 256, 0, stream>>>(
      x, nl0w, nl0b, naff, l1b, scb, l2b, tg, ws, out, N);
}
